// TileBasedRenderer_28501402977041
// MI455X (gfx1250) — matrix operand order repaired and checked by reading, not compiled, not run on hardware
//
#include <hip/hip_runtime.h>
#include <hip/hip_bf16.h>

#define G_FX 300.0f
#define G_FY 300.0f
#define G_CX 128.0f
#define G_CY 128.0f
#define G_W  256
#define G_H  256
#define G_NEAR 0.01f
#define G_FAR  100.0f
#define G_MAXR 64.0f
#define NG   512
#define CHUNK 32
#define PSTRIDE 16   // floats per gaussian param record

typedef __attribute__((ext_vector_type(16))) _Float16 v16h;
typedef __attribute__((ext_vector_type(2)))  _Float16 v2h;
typedef __attribute__((ext_vector_type(8)))  float    v8f;

// ---------------------------------------------------------------------------
// Kernel 1: per-gaussian projection / conic preprocessing
// params[n*16]: depth,u,v,ia,ibc,id,op,r,g,b,x0,x1,y0,y1,vis,pad
// ---------------------------------------------------------------------------
__global__ void preprocess_kernel(const float* __restrict__ pos,
                                  const float* __restrict__ scl,
                                  const float* __restrict__ rot,
                                  const float* __restrict__ col,
                                  const float* __restrict__ opa,
                                  const float* __restrict__ view,
                                  float* __restrict__ params)
{
    int n = blockIdx.x * blockDim.x + threadIdx.x;
    if (n >= NG) return;

    float V[16];
#pragma unroll
    for (int i = 0; i < 16; ++i) V[i] = view[i];

    float p0 = pos[n*3+0], p1 = pos[n*3+1], p2 = pos[n*3+2];
    float pc[3];
#pragma unroll
    for (int i = 0; i < 3; ++i)
        pc[i] = V[i*4+0]*p0 + V[i*4+1]*p1 + V[i*4+2]*p2 + V[i*4+3];
    float depth = -pc[2];

    // quaternion -> rotation
    float qw = rot[n*4+0], qx = rot[n*4+1], qy = rot[n*4+2], qz = rot[n*4+3];
    float qn = rsqrtf(qw*qw + qx*qx + qy*qy + qz*qz);
    qw *= qn; qx *= qn; qy *= qn; qz *= qn;
    float R[9] = {
        1.f - 2.f*(qy*qy + qz*qz), 2.f*(qx*qy - qw*qz),       2.f*(qx*qz + qw*qy),
        2.f*(qx*qy + qw*qz),       1.f - 2.f*(qx*qx + qz*qz), 2.f*(qy*qz - qw*qx),
        2.f*(qx*qz - qw*qy),       2.f*(qy*qz + qw*qx),       1.f - 2.f*(qx*qx + qy*qy)
    };
    float s[3] = { scl[n*3+0], scl[n*3+1], scl[n*3+2] };

    // RS = V[:3,:3] * R * diag(s)
    float RS[9];
#pragma unroll
    for (int i = 0; i < 3; ++i) {
#pragma unroll
        for (int l = 0; l < 3; ++l) {
            float m = 0.f;
#pragma unroll
            for (int j = 0; j < 3; ++j) m += V[i*4+j] * R[j*3+l];
            RS[i*3+l] = m * s[l];
        }
    }
    // cov3d = RS * RS^T
    float C3[9];
#pragma unroll
    for (int i = 0; i < 3; ++i) {
#pragma unroll
        for (int j = 0; j < 3; ++j) {
            float m = 0.f;
#pragma unroll
            for (int l = 0; l < 3; ++l) m += RS[i*3+l] * RS[j*3+l];
            C3[i*3+j] = m;
        }
    }

    float x = pc[0], y = pc[1], z = pc[2];
    float zs = fmaxf(fabsf(z), 0.01f);
    float zsg = z + 1e-8f;
    float sgn = (zsg > 0.f) ? 1.f : ((zsg < 0.f) ? -1.f : 0.f);
    zs *= sgn;
    float z2 = zs * zs;

    // J rows: [FX/-zs, 0, FX*x/z2], [0, FY/zs, FY*y/z2]
    float J00 = G_FX / (-zs), J02 = G_FX * x / z2;
    float J11 = G_FY / zs,    J12 = G_FY * y / z2;

    float t00 = J00*C3[0] + J02*C3[6];
    float t01 = J00*C3[1] + J02*C3[7];
    float t02 = J00*C3[2] + J02*C3[8];
    float t10 = J11*C3[3] + J12*C3[6];
    float t11 = J11*C3[4] + J12*C3[7];
    float t12 = J11*C3[5] + J12*C3[8];

    float a = t00*J00 + t02*J02;
    float b = t01*J11 + t02*J12;
    float c = t10*J00 + t12*J02;
    float d = t11*J11 + t12*J12;

    float u = G_FX * x / (-zs) + G_CX;
    float v = G_FY * y / zs + G_CY;   // FY * (-y) / (-zs) + CY

    float trace = a + d;
    float det  = fmaxf(a*d - b*c, 1e-6f);
    float disc = fmaxf(trace*trace - 4.f*det, 0.f);
    float lam  = 0.5f * (trace + sqrtf(disc));
    float radius = fminf(3.f * sqrtf(fmaxf(lam, 1e-6f)), G_MAXR);

    float vis = (depth > G_NEAR && depth < G_FAR &&
                 (u + radius) > 0.f && (u - radius) < (float)G_W &&
                 (v + radius) > 0.f && (v - radius) < (float)G_H) ? 1.f : 0.f;

    float ra = a + 1e-4f, rb = b, rc = c, rd = d + 1e-4f;
    float inv = 1.f / (ra*rd - rb*rc);
    float ia  = rd * inv;
    float ibc = (-rb - rc) * inv;
    float idd = ra * inv;

    float x0 = fmaxf(0.f, truncf(u - radius));
    float x1 = fminf((float)G_W, truncf(u + radius) + 1.f);
    float y0 = fmaxf(0.f, truncf(v - radius));
    float y1 = fminf((float)G_H, truncf(v + radius) + 1.f);

    float* P = params + n * PSTRIDE;
    P[0]  = depth; P[1] = u;  P[2] = v;
    P[3]  = ia;    P[4] = ibc; P[5] = idd;
    P[6]  = opa[n];
    P[7]  = col[n*3+0]; P[8] = col[n*3+1]; P[9] = col[n*3+2];
    P[10] = x0; P[11] = x1; P[12] = y0; P[13] = y1;
    P[14] = vis; P[15] = 0.f;
}

// ---------------------------------------------------------------------------
// Kernel 2: deterministic rank sort by depth (tie-break by index => stable)
// ---------------------------------------------------------------------------
__global__ void rank_sort_kernel(const float* __restrict__ params,
                                 int* __restrict__ order)
{
    __shared__ float keys[NG];
    int t = threadIdx.x;
    keys[t] = params[t * PSTRIDE];  // depth
    __syncthreads();
    float k = keys[t];
    int rank = 0;
    for (int m = 0; m < NG; ++m) {
        float km = keys[m];
        rank += (km < k) || (km == k && m < t);
    }
    order[rank] = t;
}

// ---------------------------------------------------------------------------
// Kernel 3: tiled alpha compositing; color reduction via V_WMMA_F32_16X16X32_F16
// Block = one 16x16 pixel tile (256 threads = 8 waves, 32 px/wave -> 2 WMMAs).
// Weights go to LDS directly in the WMMA A-fragment layout:
//   awgt[wave][grp][lane][slot], lane = mrow + 16*((k>>3)&1), slot = 8*(k>>4)+(k&7)
// Producer: one packed ds_store_b32 per (k,k+1) pair (computed address, low
// register pressure). Consumer: one 32B ds_load_b128 pair per fragment.
// ---------------------------------------------------------------------------
__global__ void __launch_bounds__(256)
render_kernel(const float* __restrict__ params,
              const int*  __restrict__ order,
              float* __restrict__ out)
{
    __shared__ float gp[CHUNK][PSTRIDE];                     // 2 KB params chunk
    __shared__ __align__(32) _Float16 awgt[8][2][32][16];    // 16 KB A fragments
    __shared__ __align__(32) _Float16 bfrag[32][16];         // 1 KB  B fragment

    const int tid  = threadIdx.x;
    const int tile = blockIdx.x;              // 0..255
    const int tx   = tile & 15, ty = tile >> 4;
    const float px = (float)(tx * 16 + (tid & 15));
    const float py = (float)(ty * 16 + (tid >> 4));
    const int lane = tid & 31;
    const int wave = tid >> 5;
    const int grp  = (tid >> 4) & 1;          // which 16-pixel group in the wave
    const int mrow = tid & 15;                // M row within the fragment

    // warm L2/WGP$ for the param + order tables: global_prefetch_b8
    __builtin_prefetch(params + tid * 2 * PSTRIDE, 0, 3);
    __builtin_prefetch(order + tid, 0, 3);

    _Float16* wrow0 = &awgt[wave][grp][mrow][0];        // K in {0-7,16-23}
    _Float16* wrow1 = &awgt[wave][grp][mrow + 16][0];   // K in {8-15,24-31}

    float T = 1.f;
    v8f acc0 = {0.f,0.f,0.f,0.f,0.f,0.f,0.f,0.f};
    v8f acc1 = {0.f,0.f,0.f,0.f,0.f,0.f,0.f,0.f};

    // per-gaussian weight: alpha * T, then advance T (sequential scan)
    auto eval_w = [&](int k) -> float {
        float u = gp[k][1], v = gp[k][2];
        float ia = gp[k][3], ibc = gp[k][4], idd = gp[k][5], op = gp[k][6];
        bool inbox = (gp[k][14] != 0.f) &
                     (px >= gp[k][10]) & (px < gp[k][11]) &
                     (py >= gp[k][12]) & (py < gp[k][13]);
        float dx = px - u, dy = py - v;
        float m = ia*dx*dx + ibc*dx*dy + idd*dy*dy;
        float alpha = inbox ? fminf(op * __expf(-0.5f * m), 0.99f) : 0.f;
        float w = alpha * T;
        T *= (1.f - alpha);
        return w;
    };

    for (int ch = 0; ch < NG / CHUNK; ++ch) {
        // stage 32 sorted gaussians (16 floats each = 512 floats) into LDS
#pragma unroll
        for (int j = 0; j < 2; ++j) {
            int lin = tid + j * 256;
            int g = lin >> 4, f = lin & 15;
            gp[g][f] = params[order[ch * CHUNK + g] * PSTRIDE + f];
        }
        __syncthreads();

        // scan 32 gaussians in pairs; pair (2k2, 2k2+1) shares a fragment
        // half-row and adjacent slots -> one packed 2xf16 ds_store_b32
#pragma unroll 4
        for (int k2 = 0; k2 < CHUNK / 2; ++k2) {
            float wa = eval_w(2 * k2);
            float wb = eval_w(2 * k2 + 1);
            v2h pk;
            pk[0] = (_Float16)wa;
            pk[1] = (_Float16)wb;
            int h    = (k2 >> 2) & 1;                       // k>>3
            int slot = ((k2 >> 3) << 3) + ((k2 & 3) << 1);  // 8*(k>>4)+(k&7)
            _Float16* base = h ? wrow1 : wrow0;
            *(v2h*)(base + slot) = pk;
        }

        // B fragment built once per block: col N = tid&15 (RGB in N<3),
        // lanes 0-15 hold K={0-7,16-23}, lanes 16-31 hold K={8-15,24-31}
        if (tid < 32) {
            int nCol = tid & 15;
            int koff = (tid >= 16) ? 8 : 0;
            v16h bv;
#pragma unroll
            for (int t2 = 0; t2 < 16; ++t2)
                bv[t2] = (nCol < 3) ? (_Float16)gp[koff + t2 + (t2 & 8)][7 + nCol]
                                    : (_Float16)0.f;
            *(v16h*)&bfrag[tid][0] = bv;
        }
        __syncthreads();

        // one 32B vector load per fragment, then the two WMMAs (uniform flow)
        v16h bf  = *(const v16h*)&bfrag[lane][0];
        v16h af0 = *(const v16h*)&awgt[wave][0][lane][0];
        v16h af1 = *(const v16h*)&awgt[wave][1][lane][0];
        acc0 = __builtin_amdgcn_wmma_f32_16x16x32_f16(
                   false, af0, false, bf, (short)0, acc0, false, false);
        acc1 = __builtin_amdgcn_wmma_f32_16x16x32_f16(
                   false, af1, false, bf, (short)0, acc1, false, false);
        __syncthreads();
    }

    // D layout writeback: lane holds col N=lane&15, VGPR r holds row M=r(+8 hi half)
    int nCol = lane & 15;
    if (nCol < 3) {
        int hi = (lane >= 16) ? 8 : 0;
#pragma unroll
        for (int g2 = 0; g2 < 2; ++g2) {
            v8f acc = g2 ? acc1 : acc0;
            int py2 = ty * 16 + wave * 2 + g2;
#pragma unroll
            for (int rr = 0; rr < 8; ++rr) {
                int px2 = tx * 16 + rr + hi;
                out[nCol * (G_W * G_H) + py2 * G_W + px2] = acc[rr];
            }
        }
    }
}

// ---------------------------------------------------------------------------
extern "C" void kernel_launch(void* const* d_in, const int* in_sizes, int n_in,
                              void* d_out, int out_size, void* d_ws, size_t ws_size,
                              hipStream_t stream)
{
    (void)in_sizes; (void)n_in; (void)out_size; (void)ws_size;
    const float* pos  = (const float*)d_in[0];
    const float* scl  = (const float*)d_in[1];
    const float* rot  = (const float*)d_in[2];
    const float* col  = (const float*)d_in[3];
    const float* opa  = (const float*)d_in[4];
    const float* view = (const float*)d_in[5];

    float* params = (float*)d_ws;                                       // 512*16 floats
    int*   order  = (int*)((char*)d_ws + NG * PSTRIDE * sizeof(float)); // 512 ints

    preprocess_kernel<<<2, 256, 0, stream>>>(pos, scl, rot, col, opa, view, params);
    rank_sort_kernel<<<1, NG, 0, stream>>>(params, order);
    render_kernel<<<256, 256, 0, stream>>>(params, order, (float*)d_out);
}
